// my_conv2_22308060136096
// MI455X (gfx1250) — matrix operand order repaired and checked by reading, not compiled, not run on hardware
//
#include <hip/hip_runtime.h>

// ---- problem constants ----
#define BATCH 32
#define CIN   192
#define HW    56
#define COUT  192
#define LPIX  (HW * HW)        // 3136 pixels per image
#define KTOT  (CIN * 9)        // 1728 reduction length

// ---- tiling ----
#define MT 64    // C_out tile per workgroup
#define NT 128   // pixel tile per workgroup = 2 batches x 64 pixels
#define KT 32    // K chunk per WMMA step
#define NCHUNK (KTOT / KT)     // 54

typedef __attribute__((ext_vector_type(16))) __bf16        bf16x16;
typedef __attribute__((ext_vector_type(8)))  float         f32x8;
typedef __attribute__((ext_vector_type(4)))  unsigned int  u32x4;

union BFrag { bf16x16 v; u32x4 q[2]; };

// ---- CDNA5 async memory->LDS path (gated; fallback = register pipeline) ----
#if __has_builtin(__builtin_amdgcn_global_load_async_to_lds_b128)
#define ASYNC_A 1
typedef int v4i_gcc __attribute__((vector_size(4 * sizeof(int))));
__device__ __forceinline__ void async_cp16(const void* g, void* l) {
  __builtin_amdgcn_global_load_async_to_lds_b128((v4i_gcc*)g, (v4i_gcc*)l, 0, 0);
}
#else
#define ASYNC_A 0
#endif

__device__ __forceinline__ void wait_async_le2() {
#if __has_builtin(__builtin_amdgcn_s_wait_asynccnt)
  __builtin_amdgcn_s_wait_asynccnt(2);
#else
  asm volatile("s_wait_asynccnt 0x2" ::: "memory");
#endif
}
__device__ __forceinline__ void wait_async_le0() {
#if __has_builtin(__builtin_amdgcn_s_wait_asynccnt)
  __builtin_amdgcn_s_wait_asynccnt(0);
#else
  asm volatile("s_wait_asynccnt 0x0" ::: "memory");
#endif
}

// ---------------------------------------------------------------------------
// Pre-pass: re-order weight K-dim to rs-major (k' = (r*3+s)*CIN + c) and split
// each fp32 into truncated bf16 hi + bf16(lo) residual. 192*1728 elements.
// ---------------------------------------------------------------------------
__global__ __launch_bounds__(256)
void pack_weights(const float* __restrict__ w,
                  unsigned short* __restrict__ wh,
                  unsigned short* __restrict__ wl) {
  int idx = blockIdx.x * 256 + threadIdx.x;     // 0 .. 192*1728-1
  int m  = idx / KTOT;
  int kp = idx - m * KTOT;
  int rs = kp / CIN;
  int c  = kp - rs * CIN;
  float v = w[m * KTOT + c * 9 + rs];
  unsigned int u = __float_as_uint(v);
  float hif = __uint_as_float(u & 0xFFFF0000u);
  float lof = v - hif;
  wh[idx] = (unsigned short)(u >> 16);
  wl[idx] = (unsigned short)(__float_as_uint(lof) >> 16);
}

// ---------------------------------------------------------------------------
// Main implicit-GEMM conv. Per WG: 64 (cout) x 128 (2 batches x 64 pix) tile,
// 8 waves, each wave a 32x32 C tile (4 v8f accs, 12 WMMA per k-chunk).
// A tile: async global->LDS double-buffered (CDNA5 path) when available.
// B tile: on-the-fly im2col, register-pipelined, truncation bf16 hi/lo split.
// ---------------------------------------------------------------------------
__global__ __launch_bounds__(256)
void conv3x3_implicit_gemm_wmma(const float* __restrict__ x,
                                const unsigned short* __restrict__ wh,
                                const unsigned short* __restrict__ wl,
                                const float* __restrict__ bias,
                                float* __restrict__ out) {
#if ASYNC_A
  __shared__ __align__(16) unsigned short sAh[2][MT * KT];   // 8 KB
  __shared__ __align__(16) unsigned short sAl[2][MT * KT];   // 8 KB
#else
  __shared__ __align__(16) unsigned short sAh[1][MT * KT];   // 4 KB
  __shared__ __align__(16) unsigned short sAl[1][MT * KT];   // 4 KB
#endif
  __shared__ __align__(16) unsigned short sBh[NT * KT];      // 8 KB
  __shared__ __align__(16) unsigned short sBl[NT * KT];      // 8 KB

  const int tid  = threadIdx.x;
  const int lane = tid & 31;
  const int wave = tid >> 5;
  const int wm   = wave >> 2;     // 0..1 : 32-row M strip
  const int wn   = wave & 3;      // 0..3 : 32-col N strip

  const int p0    = blockIdx.x * 64;   // pixel tile base (per batch)
  const int m0    = blockIdx.y * MT;   // cout tile base
  const int bpair = blockIdx.z;        // batch pair

  // ---- A staging split: 64 rows x 32 k, 8 contiguous bf16 per thread ----
  const int arow = tid >> 2;            // 0..63
  const int acol = (tid & 3) * 8;       // 0,8,16,24
  const int aoff = arow * KT + acol;
  const size_t awoff = (size_t)(m0 + arow) * KTOT + acol;

  // ---- B staging split: 128 "pixels" x 32 k, 16 channels per thread ----
  const int bu  = tid >> 1;             // 0..127
  const int btb = (tid & 1) * 16;       // 0 or 16 : channel sub-chunk
  const int ub  = bu & 63;              // pixel within 64-tile
  const int ubt = bu >> 6;              // which batch of the pair
  const int p   = p0 + ub;
  const int ho  = p / HW;
  const int wo  = p - ho * HW;
  const float* xb = x + ((size_t)(bpair * 2 + ubt)) * CIN * LPIX;

  f32x8 acc00 = {}, acc01 = {}, acc10 = {}, acc11 = {};

  float rb[16];
#if !ASYNC_A
  u32x4 ra_h, ra_l;
#endif

  int r = 0, s = 0, c0 = 0;   // coords of CURRENT chunk

  // ---- prologue: fetch chunk 0 ----
#if ASYNC_A
  async_cp16(wh + awoff, &sAh[0][aoff]);
  async_cp16(wl + awoff, &sAl[0][aoff]);
#else
  ra_h = *(const u32x4*)(wh + awoff);
  ra_l = *(const u32x4*)(wl + awoff);
#endif
  {
    int hin = ho - 1, win = wo - 1;
    bool ok = ((unsigned)hin < (unsigned)HW) && ((unsigned)win < (unsigned)HW);
    const float* xp = xb + (size_t)btb * LPIX + (hin * HW + win);
#pragma unroll
    for (int t = 0; t < 16; ++t) rb[t] = ok ? xp[(size_t)t * LPIX] : 0.0f;
  }

  for (int chunk = 0; chunk < NCHUNK; ++chunk) {
    // ---- next-chunk coords ----
    int c0n = c0 + KT, rn = r, sn = s;
    if (c0n == CIN) { c0n = 0; sn = s + 1; if (sn == 3) { sn = 0; rn = r + 1; } }
    const bool have_next = (chunk + 1 < NCHUNK);
    const size_t kbase_n = (size_t)(rn * 3 + sn) * CIN + c0n;

#if ASYNC_A
    const int buf = chunk & 1;
    // issue next A tile async into the other buffer (readers passed barrier)
    if (have_next) {
      async_cp16(wh + awoff + kbase_n, &sAh[buf ^ 1][aoff]);
      async_cp16(wl + awoff + kbase_n, &sAl[buf ^ 1][aoff]);
    }
#else
    const int buf = 0;
#endif

    // ---- commit staged B registers to LDS (truncation split) ----
#pragma unroll
    for (int t = 0; t < 16; ++t) {
      float v = rb[t];
      unsigned int u = __float_as_uint(v);
      float hif = __uint_as_float(u & 0xFFFF0000u);
      float lof = v - hif;
      sBh[bu * KT + btb + t] = (unsigned short)(u >> 16);
      sBl[bu * KT + btb + t] = (unsigned short)(__float_as_uint(lof) >> 16);
    }
#if ASYNC_A
    // current buffer's async pair must be complete (<=2 leaves only the pair
    // just issued for the next chunk in flight; async loads complete in order)
    if (have_next) wait_async_le2(); else wait_async_le0();
#else
    *(u32x4*)&sAh[0][aoff] = ra_h;
    *(u32x4*)&sAl[0][aoff] = ra_l;
#endif
    __syncthreads();

    // ---- issue next chunk's B (and fallback A) global loads ----
    if (have_next) {
#if !ASYNC_A
      ra_h = *(const u32x4*)(wh + awoff + kbase_n);
      ra_l = *(const u32x4*)(wl + awoff + kbase_n);
#endif
      int hin = ho + rn - 1, win = wo + sn - 1;
      bool ok = ((unsigned)hin < (unsigned)HW) && ((unsigned)win < (unsigned)HW);
      const float* xp = xb + (kbase_n + btb) * LPIX + (hin * HW + win);
#pragma unroll
      for (int t = 0; t < 16; ++t) rb[t] = ok ? xp[(size_t)t * LPIX] : 0.0f;
    }

    // ---- fragments (A and B both contiguous k = 16*(lane/16) + element) ----
    const unsigned short* sAhb = &sAh[buf][0];
    const unsigned short* sAlb = &sAl[buf][0];
    const int am0 = wm * 32 + (lane & 15);
    const int am1 = am0 + 16;
    const int kh  = (lane >> 4) * 16;          // A: K 0..15 vs 16..31
    BFrag ah0, al0, ah1, al1;
    ah0.q[0] = *(const u32x4*)&sAhb[am0 * KT + kh];
    ah0.q[1] = *(const u32x4*)&sAhb[am0 * KT + kh + 8];
    al0.q[0] = *(const u32x4*)&sAlb[am0 * KT + kh];
    al0.q[1] = *(const u32x4*)&sAlb[am0 * KT + kh + 8];
    ah1.q[0] = *(const u32x4*)&sAhb[am1 * KT + kh];
    ah1.q[1] = *(const u32x4*)&sAhb[am1 * KT + kh + 8];
    al1.q[0] = *(const u32x4*)&sAlb[am1 * KT + kh];
    al1.q[1] = *(const u32x4*)&sAlb[am1 * KT + kh + 8];

    const int kg = (lane >> 4) * 16;           // B: K 0..15 vs 16..31
    const int n0 = wn * 32 + (lane & 15);
    const int n1 = n0 + 16;
    BFrag bh0, bl0, bh1, bl1;
    bh0.q[0] = *(const u32x4*)&sBh[n0 * KT + kg];
    bh0.q[1] = *(const u32x4*)&sBh[n0 * KT + kg + 8];
    bl0.q[0] = *(const u32x4*)&sBl[n0 * KT + kg];
    bl0.q[1] = *(const u32x4*)&sBl[n0 * KT + kg + 8];
    bh1.q[0] = *(const u32x4*)&sBh[n1 * KT + kg];
    bh1.q[1] = *(const u32x4*)&sBh[n1 * KT + kg + 8];
    bl1.q[0] = *(const u32x4*)&sBl[n1 * KT + kg];
    bl1.q[1] = *(const u32x4*)&sBl[n1 * KT + kg + 8];

    // ---- bf16x3 split accumulation: hi*hi + hi*lo + lo*hi ----
    acc00 = __builtin_amdgcn_wmma_f32_16x16x32_bf16(false, ah0.v, false, bh0.v, (short)0, acc00, false, false);
    acc00 = __builtin_amdgcn_wmma_f32_16x16x32_bf16(false, ah0.v, false, bl0.v, (short)0, acc00, false, false);
    acc00 = __builtin_amdgcn_wmma_f32_16x16x32_bf16(false, al0.v, false, bh0.v, (short)0, acc00, false, false);
    acc01 = __builtin_amdgcn_wmma_f32_16x16x32_bf16(false, ah0.v, false, bh1.v, (short)0, acc01, false, false);
    acc01 = __builtin_amdgcn_wmma_f32_16x16x32_bf16(false, ah0.v, false, bl1.v, (short)0, acc01, false, false);
    acc01 = __builtin_amdgcn_wmma_f32_16x16x32_bf16(false, al0.v, false, bh1.v, (short)0, acc01, false, false);
    acc10 = __builtin_amdgcn_wmma_f32_16x16x32_bf16(false, ah1.v, false, bh0.v, (short)0, acc10, false, false);
    acc10 = __builtin_amdgcn_wmma_f32_16x16x32_bf16(false, ah1.v, false, bl0.v, (short)0, acc10, false, false);
    acc10 = __builtin_amdgcn_wmma_f32_16x16x32_bf16(false, al1.v, false, bh0.v, (short)0, acc10, false, false);
    acc11 = __builtin_amdgcn_wmma_f32_16x16x32_bf16(false, ah1.v, false, bh1.v, (short)0, acc11, false, false);
    acc11 = __builtin_amdgcn_wmma_f32_16x16x32_bf16(false, ah1.v, false, bl1.v, (short)0, acc11, false, false);
    acc11 = __builtin_amdgcn_wmma_f32_16x16x32_bf16(false, al1.v, false, bh1.v, (short)0, acc11, false, false);

    __syncthreads();

    r = rn; s = sn; c0 = c0n;
  }

  // ---- epilogue: C layout = VGPR e, lanes 0-15 -> M=e, 16-31 -> M=e+8 ----
  const int mrow = m0 + wm * 32 + ((lane >> 4) << 3);
  const int ncol = wn * 32 + (lane & 15);
#pragma unroll
  for (int e = 0; e < 8; ++e) {
    {
      int m = mrow + e;
      float bv = bias[m];
      int nt = ncol;
      size_t base = ((size_t)(bpair * 2 + (nt >> 6)) * COUT + m) * LPIX;
      out[base + p0 + (nt & 63)] = acc00[e] + bv;
      nt = ncol + 16;
      base = ((size_t)(bpair * 2 + (nt >> 6)) * COUT + m) * LPIX;
      out[base + p0 + (nt & 63)] = acc01[e] + bv;
    }
    {
      int m = mrow + 16 + e;
      float bv = bias[m];
      int nt = ncol;
      size_t base = ((size_t)(bpair * 2 + (nt >> 6)) * COUT + m) * LPIX;
      out[base + p0 + (nt & 63)] = acc10[e] + bv;
      nt = ncol + 16;
      base = ((size_t)(bpair * 2 + (nt >> 6)) * COUT + m) * LPIX;
      out[base + p0 + (nt & 63)] = acc11[e] + bv;
    }
  }
}

extern "C" void kernel_launch(void* const* d_in, const int* in_sizes, int n_in,
                              void* d_out, int out_size, void* d_ws, size_t ws_size,
                              hipStream_t stream) {
  (void)in_sizes; (void)n_in; (void)out_size; (void)ws_size;
  const float* x    = (const float*)d_in[0];
  const float* w    = (const float*)d_in[1];
  const float* bias = (const float*)d_in[2];
  float* out        = (float*)d_out;

  unsigned short* wh = (unsigned short*)d_ws;                 // 192*1728 ushort
  unsigned short* wl = wh + (size_t)COUT * KTOT;              // + same again

  pack_weights<<<dim3((COUT * KTOT) / 256), dim3(256), 0, stream>>>(w, wh, wl);

  dim3 grid(LPIX / 64, COUT / MT, BATCH / 2);   // (49, 3, 16)
  conv3x3_implicit_gemm_wmma<<<grid, dim3(256), 0, stream>>>(x, wh, wl, bias, out);
}
